// SpatialTemporal_68470368632977
// MI455X (gfx1250) — matrix operand order repaired and checked by reading, not compiled, not run on hardware
//
#include <hip/hip_runtime.h>

#define BATCH  16
#define TFULL  34
#define NNODE  1000
#define CINCH  32
#define DCCH   32
#define OCCH   64
#define NEDGE  16000
#define TPRIME 32          // T' = 34 - 2
#define NPAD   1024
#define NGRAPH 512         // B * T'
#define NTILES 63          // ceil(1000/16)

typedef __attribute__((ext_vector_type(16))) __bf16 bf16x16;
typedef __attribute__((ext_vector_type(8)))  float  f32x8;

typedef __attribute__((ext_vector_type(4))) unsigned int tdm_u32x4;
typedef __attribute__((ext_vector_type(8))) int          tdm_i32x8;
typedef __attribute__((ext_vector_type(4))) int          tdm_i32x4;

#if __has_builtin(__builtin_amdgcn_tensor_load_to_lds) && \
    __has_builtin(__builtin_amdgcn_s_wait_tensorcnt)
#define USE_TDM 1
#endif

__device__ __forceinline__ f32x8 zero8() {
  f32x8 z = {0.f, 0.f, 0.f, 0.f, 0.f, 0.f, 0.f, 0.f};
  return z;
}

__device__ __forceinline__ f32x8 wmma_bf16(bf16x16 a, bf16x16 b, f32x8 c) {
  return __builtin_amdgcn_wmma_f32_16x16x32_bf16(false, a, false, b, (short)0, c,
                                                 false, false);
}

// A fragment (16x32 bf16): lane holds row m = lane&15; K offsets
// {2*base..2*base+15}, base = (lane>>4)*8. Source row is f32.
__device__ __forceinline__ bf16x16 a_frag_f32(const float* __restrict__ row, int base) {
  bf16x16 f;
#pragma unroll
  for (int i = 0; i < 16; ++i) f[i] = (__bf16)row[2 * base + i];
  return f;
}

__device__ __forceinline__ bf16x16 a_frag_bf(const __bf16* __restrict__ row, int base) {
  bf16x16 f;
#pragma unroll
  for (int i = 0; i < 16; ++i) f[i] = row[2 * base + i];
  return f;
}

__device__ __forceinline__ bf16x16 zero_frag() {
  bf16x16 f;
#pragma unroll
  for (int i = 0; i < 16; ++i) f[i] = (__bf16)0.f;
  return f;
}

__device__ __forceinline__ float sigm(float x) { return 1.f / (1.f + __expf(-x)); }

// ---------------- adjacency build ----------------
__global__ void k_zero_f32(float* __restrict__ p, int n) {
  int i = blockIdx.x * 256 + threadIdx.x;
  if (i < n) p[i] = 0.f;
}

__global__ void k_deg(const int* __restrict__ ei, const float* __restrict__ ew,
                      float* __restrict__ deg) {
  int i = blockIdx.x * 256 + threadIdx.x;
  if (i < NEDGE) {
    atomicAdd(&deg[ei[NEDGE + i]], ew[i]);          // dst-indexed in-degree
  } else if (i < NEDGE + NNODE) {
    atomicAdd(&deg[i - NEDGE], 1.0f);               // self loop weight 1
  }
}

__global__ void k_dis(const float* __restrict__ deg, float* __restrict__ dis) {
  int i = blockIdx.x * 256 + threadIdx.x;
  if (i < NNODE) {
    float d = deg[i];
    dis[i] = d > 0.f ? rsqrtf(d) : 0.f;
  }
}

__global__ void k_scatter(const int* __restrict__ ei, const float* __restrict__ ew,
                          const float* __restrict__ dis, float* __restrict__ Af) {
  int i = blockIdx.x * 256 + threadIdx.x;
  if (i < NEDGE) {
    int s = ei[i], d = ei[NEDGE + i];
    atomicAdd(&Af[(size_t)d * NPAD + s], dis[s] * ew[i] * dis[d]);
  } else if (i < NEDGE + NNODE) {
    int v = i - NEDGE;
    atomicAdd(&Af[(size_t)v * NPAD + v], dis[v] * dis[v]);
  }
}

__global__ void k_cvtA(const float* __restrict__ Af, __bf16* __restrict__ Abf) {
  int i = blockIdx.x * 256 + threadIdx.x;   // exactly 1024*1024 threads
  Abf[i] = (__bf16)Af[i];
}

__global__ void k_zero_xwt_pad(__bf16* __restrict__ xwt) {
  int i = blockIdx.x * 256 + threadIdx.x;   // 512*32*16 threads
  int gc = i >> 4;
  xwt[(size_t)gc * NPAD + 1008 + (i & 15)] = (__bf16)0.f;
}

// ---------------- stage 1: gated dilated conv (WMMA 16 pos x 64 gate-out) -------------
__global__ __launch_bounds__(128) void k_conv(
    const float* __restrict__ x, const float* __restrict__ wg1,
    const float* __restrict__ bg1, const float* __restrict__ wg2,
    const float* __restrict__ bg2, float* __restrict__ out1,
    __bf16* __restrict__ gated) {
  const int lane = threadIdx.x & 31;
  const int wv = threadIdx.x >> 5;
  const int tile = blockIdx.x * 4 + wv;   // 32000 tiles exactly
  const int b = tile / (NNODE * 2);
  const int rem = tile - b * (NNODE * 2);
  const int n = rem >> 1;
  const int t0 = (rem & 1) * 16;

  const int nl = lane & 15;
  const int kh = lane >> 4;
  const int base8 = kh * 8;
  const int kb16 = kh * 16;

  // B fragments: 4 column tiles (64 gate outputs) x 2 taps. B[k=c, n] = w[n', c, 0, tap]
  bf16x16 Bf[4][2];
#pragma unroll
  for (int jt = 0; jt < 4; ++jt) {
    const float* w = (jt < 2) ? wg1 : wg2;
    const int np = (jt & 1) * 16 + nl;           // 0..31 within its gate
#pragma unroll
    for (int tap = 0; tap < 2; ++tap) {
      bf16x16 f;
#pragma unroll
      for (int i = 0; i < 16; ++i)
        f[i] = (__bf16)w[np * 64 + (kb16 + i) * 2 + tap];
      Bf[jt][tap] = f;
    }
  }

  // A fragments: row m = nl -> position t = t0 + nl; two taps (t, t+2)
  const int t = t0 + nl;
  const float* row0 = x + (((size_t)b * TFULL + t) * NNODE + n) * CINCH;
  const float* row1 = x + (((size_t)b * TFULL + t + 2) * NNODE + n) * CINCH;
  bf16x16 A0 = a_frag_f32(row0, base8);
  bf16x16 A1 = a_frag_f32(row1, base8);

  f32x8 C[4];
#pragma unroll
  for (int jt = 0; jt < 4; ++jt) C[jt] = zero8();
#pragma unroll
  for (int jt = 0; jt < 4; ++jt) {
    C[jt] = wmma_bf16(A0, Bf[jt][0], C[jt]);
    C[jt] = wmma_bf16(A1, Bf[jt][1], C[jt]);
  }

  // epilogue: tanh(g1+b1) * sigmoid(g2+b2); lane holds dc0 = nl, dc1 = 16+nl
  const int dc0 = nl, dc1 = 16 + nl;
  const float b1a = bg1[dc0], b1b = bg1[dc1];
  const float b2a = bg2[dc0], b2b = bg2[dc1];

  float v0[8], v1[8];
#pragma unroll
  for (int r = 0; r < 8; ++r) {
    v0[r] = tanhf(C[0][r] + b1a) * sigm(C[2][r] + b2a);
    v1[r] = tanhf(C[1][r] + b1b) * sigm(C[3][r] + b2b);
  }

  // out1 [B][DC][N][T']: lane writes 8 consecutive t
  float* o0 = out1 + (((size_t)b * DCCH + dc0) * NNODE + n) * TPRIME + t0 + 8 * kh;
  float* o1 = out1 + (((size_t)b * DCCH + dc1) * NNODE + n) * TPRIME + t0 + 8 * kh;
#pragma unroll
  for (int r = 0; r < 8; ++r) { o0[r] = v0[r]; o1[r] = v1[r]; }

  // gated bf16 [g=(b,t)][n][dc]  (transposed copy for the GCN stage)
#pragma unroll
  for (int r = 0; r < 8; ++r) {
    const int tt = t0 + 8 * kh + r;
    __bf16* gp = gated + (((size_t)b * TPRIME + tt) * NNODE + n) * DCCH;
    gp[dc0] = (__bf16)v0[r];
    gp[dc1] = (__bf16)v1[r];
  }
}

// ---------------- stage 2: XW = gated @ w_gcn^T, stored transposed [g][c][node] ------
__global__ __launch_bounds__(128) void k_xw(const __bf16* __restrict__ gated,
                                            const float* __restrict__ wgcn,
                                            __bf16* __restrict__ xwt) {
  const int lane = threadIdx.x & 31;
  const int wv = threadIdx.x >> 5;
  const int tile = blockIdx.x * 4 + wv;
  if (tile >= NGRAPH * NTILES) return;
  const int g = tile / NTILES;
  const int ntile = tile - g * NTILES;

  const int nl = lane & 15;
  const int kh = lane >> 4;
  const int base8 = kh * 8;
  const int kb16 = kh * 16;

  // B[k=dc, n=c] = w_gcn[c][dc]
  bf16x16 Bf[2];
#pragma unroll
  for (int ct = 0; ct < 2; ++ct) {
    const int c = ct * 16 + nl;
    bf16x16 f;
#pragma unroll
    for (int i = 0; i < 16; ++i) f[i] = (__bf16)wgcn[c * DCCH + kb16 + i];
    Bf[ct] = f;
  }

  const int node = ntile * 16 + nl;
  bf16x16 A = (node < NNODE)
                  ? a_frag_bf(gated + ((size_t)g * NNODE + node) * DCCH, base8)
                  : zero_frag();

  f32x8 C[2] = {zero8(), zero8()};
#pragma unroll
  for (int ct = 0; ct < 2; ++ct) C[ct] = wmma_bf16(A, Bf[ct], C[ct]);

  // store transposed: xwt[g][c][node], 8 consecutive nodes per lane
#pragma unroll
  for (int ct = 0; ct < 2; ++ct) {
    __bf16* dst = xwt + ((size_t)g * DCCH + ct * 16 + nl) * NPAD + ntile * 16 + 8 * kh;
#pragma unroll
    for (int r = 0; r < 8; ++r) dst[r] = (__bf16)C[ct][r];
  }
}

// ------- stage 3: gcn = A @ XW + b_gcn, fused y = gcn @ w_out^T + b_out --------------
__global__ __launch_bounds__(256) void k_gcn(const __bf16* __restrict__ Abf,
                                             const __bf16* __restrict__ xwt,
                                             const float* __restrict__ bgcn,
                                             const float* __restrict__ wout,
                                             const float* __restrict__ bout,
                                             float* __restrict__ y) {
  __shared__ __align__(32) __bf16 sXW[DCCH * NPAD];   // 64 KB: XW[g] as [c][1024]
  __shared__ float sT[8][16 * 33];                    // per-wave 16x32 transpose tile

  const int g = blockIdx.x;
  const __bf16* gsrc = xwt + (size_t)g * DCCH * NPAD;   // contiguous 64 KB block

#if defined(USE_TDM)
  // Tensor Data Mover: one DMA descriptor per block moves the 64 KB XW[g] tile
  // (16 rows x 2048 bf16, contiguous) into LDS; tracked by TENSORcnt.
  if (threadIdx.x == 0) {
    const unsigned lds_addr = (unsigned)(size_t)(&sXW[0]);   // low 32 bits = LDS offset
    const unsigned long long ga = (unsigned long long)(size_t)gsrc;
    tdm_u32x4 g0;
    g0[0] = 1u;                                          // count=1, user descriptor
    g0[1] = lds_addr;                                    // D#.lds_addr
    g0[2] = (unsigned)ga;                                // D#.global_addr[31:0]
    g0[3] = (unsigned)((ga >> 32) & 0x01FFFFFFu) | (2u << 30);  // addr[56:32], type=2
    tdm_i32x8 g1;
    g1[0] = (int)(1u << 16);       // workgroup_mask=0, data_size=1 (2 bytes)
    g1[1] = (int)(2048u << 16);    // atomic_barrier_addr=0; tensor_dim0 = 2048 (lo16)
    g1[2] = (int)(16u << 16);      // tensor_dim0 hi = 0; tensor_dim1 = 16 (lo16)
    g1[3] = (int)(2048u << 16);    // tensor_dim1 hi = 0; tile_dim0 = 2048
    g1[4] = (int)16;               // tile_dim1 = 16; tile_dim2 = 0
    g1[5] = (int)2048;             // tensor_dim0_stride lo = 2048 elements
    g1[6] = 0;                     // stride0 hi, tensor_dim1_stride lo16
    g1[7] = 0;                     // tensor_dim1_stride hi
    tdm_i32x4 gz = {0, 0, 0, 0};
#if __clang_major__ >= 23
    tdm_i32x8 gz8 = {0, 0, 0, 0, 0, 0, 0, 0};
    __builtin_amdgcn_tensor_load_to_lds(g0, g1, gz, gz, gz8, 0);
#else
    __builtin_amdgcn_tensor_load_to_lds(g0, g1, gz, gz, 0);
#endif
  }
  __builtin_amdgcn_s_wait_tensorcnt(0);   // no-op for non-issuing waves
#else
  {
    const uint4* src = (const uint4*)gsrc;
    uint4* dst = (uint4*)sXW;
    for (int i = threadIdx.x; i < 4096; i += 256) dst[i] = src[i];
  }
#endif
  __syncthreads();

  const int lane = threadIdx.x & 31;
  const int wv = threadIdx.x >> 5;
  const int nl = lane & 15;
  const int kh = lane >> 4;
  const int base8 = kh * 8;

  // out-projection weights: B[k=c, n=o] = w_out[o][c]
  bf16x16 Bout[4];
  float bo[4];
#pragma unroll
  for (int jt = 0; jt < 4; ++jt) {
    const int o = jt * 16 + nl;
    bf16x16 f;
#pragma unroll
    for (int i = 0; i < 16; ++i) f[i] = (__bf16)wout[o * CINCH + kh * 16 + i];
    Bout[jt] = f;
    bo[jt] = bout[o];
  }
  float bg[2];
#pragma unroll
  for (int ct = 0; ct < 2; ++ct) bg[ct] = bgcn[ct * 16 + nl];

  for (int itile = wv; itile < NTILES; itile += 8) {
    f32x8 C[2] = {zero8(), zero8()};
    const __bf16* arow = Abf + (size_t)(itile * 16 + nl) * NPAD;

    for (int ks = 0; ks < 32; ++ks) {
      __builtin_prefetch(arow + (ks + 4) * 32, 0, 1);
      bf16x16 A = a_frag_bf(arow + ks * 32, base8);
#pragma unroll
      for (int ct = 0; ct < 2; ++ct) {
        const bf16x16* bp =
            (const bf16x16*)(sXW + (ct * 16 + nl) * NPAD + ks * 32 + kh * 16);
        C[ct] = wmma_bf16(A, *bp, C[ct]);
      }
    }

    // add b_gcn; transpose D (lane-holds-column) -> A layout (lane-holds-row) via LDS
    float* tp = &sT[wv][0];
#pragma unroll
    for (int ct = 0; ct < 2; ++ct)
#pragma unroll
      for (int r = 0; r < 8; ++r)
        tp[(r + 8 * kh) * 33 + ct * 16 + nl] = C[ct][r] + bg[ct];
    __threadfence_block();   // order LDS writes before wave-internal readback

    bf16x16 A2;
    {
      const float* row = tp + nl * 33;
#pragma unroll
      for (int i = 0; i < 16; ++i) A2[i] = (__bf16)row[2 * base8 + i];
    }

    f32x8 D[4];
#pragma unroll
    for (int jt = 0; jt < 4; ++jt) D[jt] = zero8();
#pragma unroll
    for (int jt = 0; jt < 4; ++jt) D[jt] = wmma_bf16(A2, Bout[jt], D[jt]);

    // y [G][N][64]
#pragma unroll
    for (int jt = 0; jt < 4; ++jt) {
      const int o = jt * 16 + nl;
#pragma unroll
      for (int r = 0; r < 8; ++r) {
        const int i = itile * 16 + r + 8 * kh;
        if (i < NNODE) y[((size_t)g * NNODE + i) * OCCH + o] = D[jt][r] + bo[jt];
      }
    }
    __threadfence_block();
  }
}

extern "C" void kernel_launch(void* const* d_in, const int* in_sizes, int n_in,
                              void* d_out, int out_size, void* d_ws, size_t ws_size,
                              hipStream_t stream) {
  const float* x    = (const float*)d_in[0];
  const int*   ei   = (const int*)d_in[1];
  const float* ew   = (const float*)d_in[2];
  const float* wg1  = (const float*)d_in[3];
  const float* bg1  = (const float*)d_in[4];
  const float* wg2  = (const float*)d_in[5];
  const float* bg2  = (const float*)d_in[6];
  const float* wgcn = (const float*)d_in[7];
  const float* bgcn = (const float*)d_in[8];
  const float* wout = (const float*)d_in[9];
  const float* bout = (const float*)d_in[10];

  float* ws_f = (float*)d_ws;
  float* deg = ws_f;                        // [1024] f32
  float* dis = ws_f + 1024;                 // [1024] f32
  float* Af  = ws_f + 4096;                 // [1024*1024] f32 dense A
  __bf16* Abf   = (__bf16*)(ws_f + 4096 + 1048576);                      // 2 MB
  __bf16* gated = (__bf16*)(ws_f + 4096 + 1048576 + 524288);             // 32 MB
  __bf16* xwt   = (__bf16*)(ws_f + 4096 + 1048576 + 524288 + 8192000);   // 32 MB

  float* out1 = (float*)d_out;
  float* yout = out1 + (size_t)BATCH * DCCH * NNODE * TPRIME;

  k_zero_f32<<<(1052672 + 255) / 256, 256, 0, stream>>>(ws_f, 1052672);
  k_deg<<<(NEDGE + NNODE + 255) / 256, 256, 0, stream>>>(ei, ew, deg);
  k_dis<<<(NNODE + 255) / 256, 256, 0, stream>>>(deg, dis);
  k_scatter<<<(NEDGE + NNODE + 255) / 256, 256, 0, stream>>>(ei, ew, dis, Af);
  k_cvtA<<<4096, 256, 0, stream>>>(Af, Abf);
  k_zero_xwt_pad<<<1024, 256, 0, stream>>>(xwt);
  k_conv<<<8000, 128, 0, stream>>>(x, wg1, bg1, wg2, bg2, out1, gated);
  k_xw<<<8064, 128, 0, stream>>>(gated, wgcn, xwt);
  k_gcn<<<512, 256, 0, stream>>>(Abf, xwt, bgcn, wout, bout, yout);
}
